// MultiHeadAttention_58634893525856
// MI455X (gfx1250) — matrix operand order repaired and checked by reading, not compiled, not run on hardware
//
#include <hip/hip_runtime.h>
#include <hip/hip_bf16.h>

// ---------------------------------------------------------------------------
// MHA forward for gfx1250 (MI455X): bf16 WMMA + TDM (tensor_load_to_lds)
// double-buffered flash attention.  B=2, S=2048, D_MODEL=1024, H=16, dk=64.
// ---------------------------------------------------------------------------

typedef __bf16 bf16_t;
typedef __attribute__((ext_vector_type(16))) __bf16 v16bf;
typedef __attribute__((ext_vector_type(8)))  __bf16 v8bf;
typedef __attribute__((ext_vector_type(8)))  float  v8f;
typedef __attribute__((ext_vector_type(4)))  float  v4f;
typedef __attribute__((ext_vector_type(4)))  unsigned int v4u;
typedef __attribute__((ext_vector_type(8)))  int v8i;
typedef __attribute__((ext_vector_type(4)))  int v4i;

#define WMMA_BF16(A, B, C) \
  __builtin_amdgcn_wmma_f32_16x16x32_bf16(false, (A), false, (B), (short)0, (C), false, false)

static constexpr int S_LEN   = 2048;
static constexpr int D_MODEL = 1024;
static constexpr int H_NUM   = 16;
static constexpr int D_K     = 64;
static constexpr int M_ROWS  = 2 * S_LEN;   // 4096

// --- A fragment (16x32 bf16), row-major source with leading dim `ld`.
// lanes 0-15: M=lane,    K = {0..7, 16..23}
// lanes16-31: M=lane-16, K = {8..15, 24..31}
__device__ inline v16bf load_a_bf16(const bf16_t* __restrict__ base, int ld) {
  const int lane = threadIdx.x & 31;
  const int m    = lane & 15;
  const int ko   = (lane >> 4) << 3;   // 0 or 8
  const bf16_t* r = base + m * ld;
  v8bf lo = *(const v8bf*)(r + ko);
  v8bf hi = *(const v8bf*)(r + ko + 16);
  v16bf a;
#pragma unroll
  for (int i = 0; i < 8; ++i) { a[i] = lo[i]; a[i + 8] = hi[i]; }
  return a;
}

// --- A fragment from an fp32 row-major matrix (convert to bf16 on the fly).
__device__ inline v16bf load_a_f32(const float* __restrict__ base, int ld) {
  const int lane = threadIdx.x & 31;
  const int m    = lane & 15;
  const int ko   = (lane >> 4) << 3;   // 0 or 8
  const float* r = base + m * ld;
  v4f x0 = *(const v4f*)(r + ko);
  v4f x1 = *(const v4f*)(r + ko + 4);
  v4f x2 = *(const v4f*)(r + ko + 16);
  v4f x3 = *(const v4f*)(r + ko + 20);
  v16bf a;
#pragma unroll
  for (int i = 0; i < 4; ++i) {
    a[i]      = (bf16_t)x0[i];
    a[i + 4]  = (bf16_t)x1[i];
    a[i + 8]  = (bf16_t)x2[i];
    a[i + 12] = (bf16_t)x3[i];
  }
  return a;
}

// --- B fragment (32x16 bf16): memory holds N rows contiguous over K
// (N x K row-major source; exactly W for y = x @ W^T).
// lanes 0-15: N=lane, K = {0..7, 16..23} ; lanes16-31: N=lane-16, K = {8..15, 24..31}
__device__ inline v16bf load_b_nk(const bf16_t* __restrict__ base, int ld) {
  const int lane = threadIdx.x & 31;
  const int n    = lane & 15;
  const int ko   = (lane >> 4) << 3;   // 0 or 8
  const bf16_t* r = base + n * ld + ko;
  v8bf lo = *(const v8bf*)(r);
  v8bf hi = *(const v8bf*)(r + 16);
  v16bf b;
#pragma unroll
  for (int i = 0; i < 8; ++i) { b[i] = lo[i]; b[i + 8] = hi[i]; }
  return b;
}

// ---------------------------------------------------------------------------
// TDM: 2D tile load Global -> LDS (bf16 elements).  D# per ISA 08 §8.
// All operands wave-uniform -> SGPR descriptor groups.  Groups 2/3 = 0 (2D).
// This toolchain's builtin takes 6 args (extra int32x8 group before cpol).
// ---------------------------------------------------------------------------
__device__ inline void tdm_load_2d(unsigned lds_addr, const bf16_t* gptr,
                                   unsigned tensor_d0, unsigned tensor_d1,
                                   unsigned tile_d0, unsigned tile_d1,
                                   unsigned long d0_stride) {
  const unsigned long ga = (unsigned long)(uintptr_t)gptr;
  v4u g0;
  g0[0] = 1u;                                           // count=1, user D#
  g0[1] = lds_addr;                                     // LDS byte address
  g0[2] = (unsigned)(ga & 0xFFFFFFFFu);                 // global_addr[31:0]
  g0[3] = (unsigned)((ga >> 32) & 0x1FFFFFFu) | (2u << 30);  // [56:32] | type=2
  v8i g1;
  g1[0] = (int)(1u << 16);                              // data_size = 2 bytes
  g1[1] = (int)((tensor_d0 & 0xFFFFu) << 16);           // tensor_dim0[15:0]
  g1[2] = (int)((tensor_d0 >> 16) | ((tensor_d1 & 0xFFFFu) << 16));
  g1[3] = (int)((tensor_d1 >> 16) | (tile_d0 << 16));   // tile_dim0
  g1[4] = (int)(tile_d1 & 0xFFFFu);                     // tile_dim1, tile_dim2=0
  g1[5] = (int)(d0_stride & 0xFFFFFFFFu);               // dim0_stride[31:0]
  g1[6] = (int)((d0_stride >> 32) & 0xFFFFu);           // dim0_stride[47:32]
  g1[7] = 0;                                            // dim1_stride unused (2D)
  const v4i z4 = {0, 0, 0, 0};
  const v8i z8 = {0, 0, 0, 0, 0, 0, 0, 0};
  __builtin_amdgcn_tensor_load_to_lds(g0, g1, z4, z4, z8, 0);
}

// ---------------------------------------------------------------------------
// Kernel 0: fp32 -> bf16 weight conversion.
// ---------------------------------------------------------------------------
__global__ void cvt_bf16_kernel(const float* __restrict__ src,
                                bf16_t* __restrict__ dst, int n) {
  int i = blockIdx.x * blockDim.x + threadIdx.x;
  if (i < n) dst[i] = (bf16_t)src[i];
}

// ---------------------------------------------------------------------------
// Kernel 1: projection  Y = (X @ W^T + bias) * scale, output bf16.
// mode 0: head-split [B,H,S,dk] ; mode 2: transposed-V [B,H,dk,S].
// Each wave: 16x64 output strip (A-frag reused across 4 B-frags / 4 WMMAs).
// ---------------------------------------------------------------------------
__global__ __launch_bounds__(256)
void proj_kernel(const float* __restrict__ X, const bf16_t* __restrict__ W,
                 const float* __restrict__ bias, bf16_t* __restrict__ Y,
                 int mode, float scale) {
  const int wave = threadIdx.x >> 5;
  const int lane = threadIdx.x & 31;
  const int m0 = blockIdx.x * 16;
  const int n0 = (blockIdx.y * 8 + wave) * 64;

  v8f acc[4];
#pragma unroll
  for (int j = 0; j < 4; ++j) acc[j] = v8f{0.f,0.f,0.f,0.f,0.f,0.f,0.f,0.f};

#pragma unroll 2
  for (int kk = 0; kk < D_MODEL; kk += 32) {
    v16bf a = load_a_f32(X + (long)m0 * D_MODEL + kk, D_MODEL);
#pragma unroll
    for (int j = 0; j < 4; ++j) {
      v16bf b = load_b_nk(W + (long)(n0 + j * 16) * D_MODEL + kk, D_MODEL);
      acc[j] = WMMA_BF16(a, b, acc[j]);
    }
  }

  const int rbase = (lane >> 4) << 3;
#pragma unroll
  for (int j = 0; j < 4; ++j) {
    const int n = n0 + j * 16 + (lane & 15);
    const float bn = bias[n];
    const int h = n >> 6, d = n & 63;
#pragma unroll
    for (int r = 0; r < 8; ++r) {
      const int m = m0 + rbase + r;
      const int bi = m >> 11, s = m & (S_LEN - 1);
      const float y = (acc[j][r] + bn) * scale;
      long idx;
      if (mode == 2) idx = (((long)(bi * H_NUM + h) * D_K + d) * S_LEN + s);
      else           idx = (((long)(bi * H_NUM + h) * S_LEN + s) * D_K + d);
      Y[idx] = (bf16_t)y;
    }
  }
}

// ---------------------------------------------------------------------------
// Kernel 2: flash attention with TDM double-buffered K/V staging in LDS.
// Per wave: 16 query rows, dk=64 accumulator in VGPRs, 32-key chunks with
// online softmax.  Qh/Kh: [B,H,S,dk] bf16 (Q pre-scaled by 1/sqrt(dk));
// VT: [B,H,dk,S] bf16.  O: [B,S,H*dk] bf16.
// ---------------------------------------------------------------------------
__global__ __launch_bounds__(256)
void flash_kernel(const bf16_t* __restrict__ Qh, const bf16_t* __restrict__ Kh,
                  const bf16_t* __restrict__ VT, bf16_t* __restrict__ O) {
  __shared__ __align__(16) bf16_t kbuf[2][32][64];   // keys x dk      (8 KB)
  __shared__ __align__(16) bf16_t vbuf[2][64][32];   // dk x keys      (8 KB)
  __shared__ __align__(16) bf16_t pbuf[8][16][40];   // per-wave P tile (10 KB)

  const int wave = threadIdx.x >> 5;
  const int lane = threadIdx.x & 31;
  const int bh = blockIdx.y;                  // b*16 + h
  const int q0 = blockIdx.x * 128 + wave * 16;

  const bf16_t* Qb = Qh + (long)bh * S_LEN * D_K + (long)q0 * D_K;
  const bf16_t* Kb = Kh + (long)bh * S_LEN * D_K;
  const bf16_t* Vb = VT + (long)bh * D_K * S_LEN;

  // Persistent Q fragments (dk = 64 -> two K=32 steps).
  const v16bf qa0 = load_a_bf16(Qb, D_K);
  const v16bf qa1 = load_a_bf16(Qb + 32, D_K);

  v8f acc0 = {0.f,0.f,0.f,0.f,0.f,0.f,0.f,0.f};
  v8f acc1 = acc0, acc2 = acc0, acc3 = acc0;
  float mrow[8], lrow[8];
#pragma unroll
  for (int r = 0; r < 8; ++r) { mrow[r] = -1.0e30f; lrow[r] = 0.f; }

  const int nchunk = S_LEN / 32;              // 64

  // Prologue: wave 0 posts the chunk-0 TDM pair.
  if (threadIdx.x < 32) {
    tdm_load_2d((unsigned)(uintptr_t)&kbuf[0][0][0], Kb,
                D_K, S_LEN, D_K, 32, D_K);
    tdm_load_2d((unsigned)(uintptr_t)&vbuf[0][0][0], Vb,
                S_LEN, D_K, 32, D_K, S_LEN);
  }

  for (int i = 0; i < nchunk; ++i) {
    const int buf = i & 1;

    // Pipeline: post chunk i+1, then require chunk i's pair retired.
    if (threadIdx.x < 32) {
      if (i + 1 < nchunk) {
        const int kn = (i + 1) * 32;
        tdm_load_2d((unsigned)(uintptr_t)&kbuf[buf ^ 1][0][0],
                    Kb + (long)kn * D_K, D_K, S_LEN, D_K, 32, D_K);
        tdm_load_2d((unsigned)(uintptr_t)&vbuf[buf ^ 1][0][0],
                    Vb + kn, S_LEN, D_K, 32, D_K, S_LEN);
        __builtin_amdgcn_s_wait_tensorcnt(2);   // chunk i done, i+1 in flight
      } else {
        __builtin_amdgcn_s_wait_tensorcnt(0);   // last chunk fully landed
      }
    }
    __syncthreads();                            // publish kbuf/vbuf[buf]

    // ---- scores: 16 queries x 32 keys (two 16x16 tiles), dk in 2 K-steps
    v8f s0 = {0.f,0.f,0.f,0.f,0.f,0.f,0.f,0.f};
    v8f s1 = s0;
    {
      v16bf b00 = load_b_nk(&kbuf[buf][0][0],  D_K);
      v16bf b01 = load_b_nk(&kbuf[buf][16][0], D_K);
      s0 = WMMA_BF16(qa0, b00, s0);
      s1 = WMMA_BF16(qa0, b01, s1);
      v16bf b10 = load_b_nk(&kbuf[buf][0][32],  D_K);
      v16bf b11 = load_b_nk(&kbuf[buf][16][32], D_K);
      s0 = WMMA_BF16(qa1, b10, s0);
      s1 = WMMA_BF16(qa1, b11, s1);
    }

    // ---- online softmax per row (rows live across 16-lane halves)
    const int rowbase = (lane >> 4) << 3;
    const int col     = lane & 15;
#pragma unroll
    for (int r = 0; r < 8; ++r) {
      float v0 = s0[r], v1 = s1[r];
      float mx = fmaxf(v0, v1);
      mx = fmaxf(mx, __shfl_xor(mx, 1, 32));
      mx = fmaxf(mx, __shfl_xor(mx, 2, 32));
      mx = fmaxf(mx, __shfl_xor(mx, 4, 32));
      mx = fmaxf(mx, __shfl_xor(mx, 8, 32));
      const float mnew = fmaxf(mrow[r], mx);
      const float p0 = __expf(v0 - mnew);
      const float p1 = __expf(v1 - mnew);
      const float corr = __expf(mrow[r] - mnew);
      float ps = p0 + p1;
      ps += __shfl_xor(ps, 1, 32);
      ps += __shfl_xor(ps, 2, 32);
      ps += __shfl_xor(ps, 4, 32);
      ps += __shfl_xor(ps, 8, 32);
      lrow[r] = lrow[r] * corr + ps;
      mrow[r] = mnew;
      acc0[r] *= corr; acc1[r] *= corr; acc2[r] *= corr; acc3[r] *= corr;
      // C-layout -> LDS row-major P tile (re-loaded as A fragment)
      pbuf[wave][rowbase + r][col]      = (bf16_t)p0;
      pbuf[wave][rowbase + r][col + 16] = (bf16_t)p1;
    }
    asm volatile("s_wait_dscnt 0" ::: "memory");   // same-wave LDS RAW

    // ---- PV: one A fragment (16 q x 32 keys), four d-tiles of V^T from LDS
    v16bf pa  = load_a_bf16(&pbuf[wave][0][0], 40);
    v16bf vb0 = load_b_nk(&vbuf[buf][0][0],  32);
    v16bf vb1 = load_b_nk(&vbuf[buf][16][0], 32);
    v16bf vb2 = load_b_nk(&vbuf[buf][32][0], 32);
    v16bf vb3 = load_b_nk(&vbuf[buf][48][0], 32);
    acc0 = WMMA_BF16(pa, vb0, acc0);
    acc1 = WMMA_BF16(pa, vb1, acc1);
    acc2 = WMMA_BF16(pa, vb2, acc2);
    acc3 = WMMA_BF16(pa, vb3, acc3);

    __syncthreads();                 // all waves done with buf before overwrite
  }

  // ---- epilogue: normalize and store merged-head bf16 [B,S,H*dk]
  const int col   = lane & 15;
  const int rbase = (lane >> 4) << 3;
  const int bi = bh >> 4, h = bh & 15;
#pragma unroll
  for (int r = 0; r < 8; ++r) {
    const float inv = 1.0f / lrow[r];
    const int q = q0 + rbase + r;
    const long base = ((long)(bi * S_LEN + q) * D_MODEL) + h * D_K;
    O[base + 0  + col] = (bf16_t)(acc0[r] * inv);
    O[base + 16 + col] = (bf16_t)(acc1[r] * inv);
    O[base + 32 + col] = (bf16_t)(acc2[r] * inv);
    O[base + 48 + col] = (bf16_t)(acc3[r] * inv);
  }
}

// ---------------------------------------------------------------------------
// Kernel 3: output projection  out = Xb @ Wo^T + bo, fp32 output.
// Each wave: 16x64 strip, 4 accumulators.
// ---------------------------------------------------------------------------
__global__ __launch_bounds__(256)
void oproj_kernel(const bf16_t* __restrict__ Xb, const bf16_t* __restrict__ W,
                  const float* __restrict__ bias, float* __restrict__ Yout) {
  const int wave = threadIdx.x >> 5;
  const int lane = threadIdx.x & 31;
  const int m0 = blockIdx.x * 16;
  const int n0 = (blockIdx.y * 8 + wave) * 64;

  v8f acc[4];
#pragma unroll
  for (int j = 0; j < 4; ++j) acc[j] = v8f{0.f,0.f,0.f,0.f,0.f,0.f,0.f,0.f};

#pragma unroll 2
  for (int kk = 0; kk < D_MODEL; kk += 32) {
    v16bf a = load_a_bf16(Xb + (long)m0 * D_MODEL + kk, D_MODEL);
#pragma unroll
    for (int j = 0; j < 4; ++j) {
      v16bf b = load_b_nk(W + (long)(n0 + j * 16) * D_MODEL + kk, D_MODEL);
      acc[j] = WMMA_BF16(a, b, acc[j]);
    }
  }

  const int rbase = (lane >> 4) << 3;
#pragma unroll
  for (int j = 0; j < 4; ++j) {
    const int n = n0 + j * 16 + (lane & 15);
    const float bn = bias[n];
#pragma unroll
    for (int r = 0; r < 8; ++r) {
      const int m = m0 + rbase + r;
      Yout[(long)m * D_MODEL + n] = acc[j][r] + bn;
    }
  }
}

// ---------------------------------------------------------------------------
// Host launch
// ---------------------------------------------------------------------------
extern "C" void kernel_launch(void* const* d_in, const int* in_sizes, int n_in,
                              void* d_out, int out_size, void* d_ws, size_t ws_size,
                              hipStream_t stream) {
  const float* Q  = (const float*)d_in[0];
  const float* K  = (const float*)d_in[1];
  const float* V  = (const float*)d_in[2];
  const float* Wq = (const float*)d_in[3];
  const float* bq = (const float*)d_in[4];
  const float* Wk = (const float*)d_in[5];
  const float* bk = (const float*)d_in[6];
  const float* Wv = (const float*)d_in[7];
  const float* bv = (const float*)d_in[8];
  const float* Wo = (const float*)d_in[9];
  const float* bo = (const float*)d_in[10];

  char* ws = (char*)d_ws;
  bf16_t* wqb = (bf16_t*)(ws + (0ul  << 20));   // 2 MB each
  bf16_t* wkb = (bf16_t*)(ws + (2ul  << 20));
  bf16_t* wvb = (bf16_t*)(ws + (4ul  << 20));
  bf16_t* wob = (bf16_t*)(ws + (6ul  << 20));
  bf16_t* qh  = (bf16_t*)(ws + (8ul  << 20));   // 8 MB each
  bf16_t* kh  = (bf16_t*)(ws + (16ul << 20));
  bf16_t* vT  = (bf16_t*)(ws + (24ul << 20));
  bf16_t* ah  = (bf16_t*)(ws + (32ul << 20));

  const int NW = D_MODEL * D_MODEL;             // 1M elements per weight
  cvt_bf16_kernel<<<NW / 256, 256, 0, stream>>>(Wq, wqb, NW);
  cvt_bf16_kernel<<<NW / 256, 256, 0, stream>>>(Wk, wkb, NW);
  cvt_bf16_kernel<<<NW / 256, 256, 0, stream>>>(Wv, wvb, NW);
  cvt_bf16_kernel<<<NW / 256, 256, 0, stream>>>(Wo, wob, NW);

  dim3 pg(M_ROWS / 16, D_MODEL / 512);          // (256, 2): 16x64 strip per wave
  proj_kernel<<<pg, 256, 0, stream>>>(Q, wqb, bq, qh, /*mode=*/0, /*scale=*/0.125f);
  proj_kernel<<<pg, 256, 0, stream>>>(K, wkb, bk, kh, /*mode=*/0, /*scale=*/1.0f);
  proj_kernel<<<pg, 256, 0, stream>>>(V, wvb, bv, vT, /*mode=*/2, /*scale=*/1.0f);

  flash_kernel<<<dim3(S_LEN / 128, 2 * H_NUM), 256, 0, stream>>>(qh, kh, vT, ah);

  oproj_kernel<<<pg, 256, 0, stream>>>(ah, wob, bo, (float*)d_out);
}
